// SignalDecoder_48241072668823
// MI455X (gfx1250) — matrix operand order repaired and checked by reading, not compiled, not run on hardware
//
#include <hip/hip_runtime.h>

// ---------------------------------------------------------------------------
// Bidirectional 2-layer LSTM decoder, 52 sequential steps.
// B=1024, H=256, NFEAT=512, K = 512(Wih)+256(Whh) = 768 fused.
// fp16 WMMA (v_wmma_f32_16x16x32_f16), f32 accumulation, double-buffered LDS,
// CDNA5 async global->LDS staging (GLOBAL_LOAD_ASYNC_TO_LDS_B128 / ASYNCcnt).
// ---------------------------------------------------------------------------

#define B_    1024
#define HID_  256
#define NF_   512
#define KTOT  768
#define NSTEP 52
#define BH    (B_ * HID_)          // 262144 elements per (cell) state plane

#define XS 40    // LDS row stride (halves) for X tile rows of 32 (80B = 20 banks)
#define WS 40    // LDS row stride (halves) for W tile rows of 32
#define GS 132   // LDS row stride (floats) for 64x128 gate tile (conflict pad)

#if defined(__has_builtin)
#if __has_builtin(__builtin_amdgcn_global_load_async_to_lds_b128) && \
    __has_builtin(__builtin_amdgcn_s_wait_asynccnt)
#define USE_ASYNC 1
#endif
#endif
#ifndef USE_ASYNC
#define USE_ASYNC 0
#endif

#define AS1 __attribute__((address_space(1)))
#define AS3 __attribute__((address_space(3)))

typedef int i4v __attribute__((vector_size(16)));   // matches builtin's V4i

typedef __attribute__((ext_vector_type(16))) _Float16 v16h;
typedef __attribute__((ext_vector_type(8)))  float    v8f;

union Frag16 { v16h v; uint4 q[2]; };

union alignas(16) SharedU {
  struct { _Float16 X[2][64 * XS]; _Float16 W[2][128 * WS]; } s; // ~30 KB
  float G[64 * GS];                                              // ~33 KB
};

__device__ __forceinline__ float sigf(float x) {
  return 1.0f / (1.0f + __expf(-x));
}

// ---------------------------------------------------------------------------
// One LSTM layer (both directions) for one time step.
// Block: 256 threads (8 waves); tile: 64 batch rows x 32 h-cols
//   => 64 x 128 gate tile (strips at n, 256+n, 512+n, 768+n).
// Wave w: M-subtile w%4, N-group w/4 => 4 WMMA accumulators; K-loop 24 x 32,
// software-pipelined with double-buffered LDS (1 barrier / iteration).
// ---------------------------------------------------------------------------
__global__ __launch_bounds__(256) void lstm_layer_kernel(
    const _Float16* __restrict__ x0,     // Xcat cols   0..255   [B,256]
    const _Float16* __restrict__ x1,     // Xcat cols 256..511   [B,256]
    const _Float16* __restrict__ hprev,  // + d*BH -> Xcat cols 512..767
    _Float16* __restrict__ hout,         // + d*BH -> new h (fp16)
    float*    __restrict__ cstate,       // + d*BH -> c (fp32, in-place)
    const _Float16* __restrict__ Wl,     // [2][1024][768] fp16
    const float*    __restrict__ biasL,  // [2][1024] (b_ih + b_hh)
    float* __restrict__ y)               // d_out + t*NF_ , or nullptr (layer0)
{
  __shared__ SharedU lds;

  const int tid = threadIdx.x;
  const int m0  = blockIdx.x * 64;   // batch tile
  const int n0  = blockIdx.y * 32;   // h-column tile
  const int d   = blockIdx.z;        // direction

  const _Float16* x2 = hprev + (size_t)d * BH;          // stays in SGPRs
  const _Float16* Wd = Wl + (size_t)d * 1024 * KTOT;

  const int w     = tid >> 5;
  const int ln    = tid & 31;
  const int msub  = w & 3;
  const int ngrp  = w >> 2;
  const int lhalf = ln >> 4;
  const int l16   = ln & 15;

  // per-thread staging coordinates (constant across iterations)
  const int xr = tid >> 2;           // X tile row 0..63
  const int xc = (tid & 3) * 8;      // X tile col chunk

  auto stage = [&](int itS, int bufS) {
    const int k0 = itS * 32;
    const int ko = k0 & 255;
    // uniform SGPR select => global_load ... saddr (no flat)
    const _Float16* xsrc = (k0 < 256) ? x0 : (k0 < 512) ? x1 : x2;
    {
      const _Float16* g = xsrc + (size_t)(m0 + xr) * 256 + ko + xc;
      _Float16* l = &lds.s.X[bufS][xr * XS + xc];
#if USE_ASYNC
      __builtin_amdgcn_global_load_async_to_lds_b128(
          (AS1 i4v*)g, (AS3 i4v*)l, 0, 0);
#else
      *(uint4*)l = *(const uint4*)g;
#endif
    }
#pragma unroll
    for (int h = 0; h < 2; ++h) {
      const int li   = tid + h * 256;
      const int lr   = li >> 2;            // W tile row 0..127
      const int q    = (li & 3) * 8;       // col chunk
      const int gate = lr >> 5;
      const int nn   = lr & 31;
      const _Float16* g = Wd + (size_t)(gate * 256 + n0 + nn) * KTOT + k0 + q;
      _Float16* l = &lds.s.W[bufS][lr * WS + q];
#if USE_ASYNC
      __builtin_amdgcn_global_load_async_to_lds_b128(
          (AS1 i4v*)g, (AS3 i4v*)l, 0, 0);
#else
      *(uint4*)l = *(const uint4*)g;
#endif
    }
  };

  const v8f z8 = {0.f, 0.f, 0.f, 0.f, 0.f, 0.f, 0.f, 0.f};
  v8f acc[4] = {z8, z8, z8, z8};

  // prologue: stage iteration 0 into buffer 0
  stage(0, 0);
#if USE_ASYNC
  __builtin_amdgcn_s_wait_asynccnt(0);
#endif
  __syncthreads();

  for (int it = 0; it < 24; ++it) {
    const int cur = it & 1;
    if (it + 1 < 24) stage(it + 1, cur ^ 1);   // prefetch next K-slab

    const _Float16* Xb = &lds.s.X[cur][0];
    const _Float16* Wb = &lds.s.W[cur][0];

    // A fragment (16x32 f16): lanes 0-15 K{0..7,16..23}, 16-31 K{8..15,24..31}
    Frag16 a;
    {
      const int m    = msub * 16 + l16;
      const int base = lhalf * 8;
      a.q[0] = *(const uint4*)&Xb[m * XS + base];
      a.q[1] = *(const uint4*)&Xb[m * XS + base + 16];
    }
    // Load ALL four B fragments into distinct registers first so the ds_loads
    // overlap, then issue the four WMMAs back-to-back (avoids a full
    // s_wait_dscnt 0 before every WMMA from register coalescing).
    Frag16 b[4];
#pragma unroll
    for (int j = 0; j < 4; ++j) {
      const int row  = ngrp * 64 + j * 16 + l16;
      const int koff = lhalf * 8;
      b[j].q[0] = *(const uint4*)&Wb[row * WS + koff];
      b[j].q[1] = *(const uint4*)&Wb[row * WS + koff + 16];
    }
#pragma unroll
    for (int j = 0; j < 4; ++j) {
      acc[j] = __builtin_amdgcn_wmma_f32_16x16x32_f16(
          false, a.v, false, b[j].v, (short)0, acc[j], false, false);
    }

#if USE_ASYNC
    __builtin_amdgcn_s_wait_asynccnt(0);
#endif
    __syncthreads();   // one barrier per iteration (double-buffered)
  }

  // --- spill D tiles to LDS gate buffer [64][GS] (f32) ---
#pragma unroll
  for (int j = 0; j < 4; ++j) {
#pragma unroll
    for (int v = 0; v < 8; ++v) {
      const int r   = msub * 16 + v + 8 * lhalf;   // C/D layout: M = v (+8)
      const int col = ngrp * 64 + j * 16 + l16;    // N = lane%16
      lds.G[r * GS + col] = acc[j][v];
    }
  }
  __syncthreads();

  // --- fused LSTM cell epilogue: 2048 (b,h) outputs, 8 per thread ---
  float*     cptr = cstate + (size_t)d * BH;
  _Float16*  hptr = hout   + (size_t)d * BH;
  const float* bi = biasL + d * 1024;

#pragma unroll
  for (int e = 0; e < 8; ++e) {
    const int jj  = tid + e * 256;
    const int r   = jj >> 5;
    const int col = jj & 31;
    const int nc  = n0 + col;

    const float gi = lds.G[r * GS + col]      + bi[nc];
    const float gf = lds.G[r * GS + 32 + col] + bi[256 + nc];
    const float gg = lds.G[r * GS + 64 + col] + bi[512 + nc];
    const float go = lds.G[r * GS + 96 + col] + bi[768 + nc];

    const size_t sidx = (size_t)(m0 + r) * 256 + nc;
    const float c_old = cptr[sidx];

    const float i_ = sigf(gi);
    const float f_ = sigf(gf);
    const float g_ = tanhf(gg);
    const float o_ = sigf(go);
    const float c2 = f_ * c_old + i_ * g_;
    const float h2 = o_ * tanhf(c2);

    cptr[sidx] = c2;
    hptr[sidx] = (_Float16)h2;
    if (y)  // layer 1: y[:, t, d*256 + nc]
      y[(size_t)(m0 + r) * (NSTEP * NF_) + d * HID_ + nc] = h2;
  }
}

// ---------------------------------------------------------------------------
// Prep kernels
// ---------------------------------------------------------------------------
__global__ void prep_w_kernel(const float* __restrict__ Wih,   // [2,1024,512]
                              const float* __restrict__ Whh,   // [2,1024,256]
                              _Float16* __restrict__ out) {    // [2,1024,768]
  const int idx = blockIdx.x * 256 + threadIdx.x;  // 2*1024*768 total
  const int dn  = idx / KTOT;
  const int k   = idx % KTOT;
  const float v = (k < 512) ? Wih[(size_t)dn * 512 + k]
                            : Whh[(size_t)dn * 256 + (k - 512)];
  out[idx] = (_Float16)v;
}

__global__ void prep_b_kernel(const float* __restrict__ bih,
                              const float* __restrict__ bhh,
                              float* __restrict__ out) {
  const int idx = blockIdx.x * 256 + threadIdx.x;  // 2048 total
  out[idx] = bih[idx] + bhh[idx];
}

// code[b, 2, 4, 256] -> h0 (fp16, cell-major) and c0 (fp32, cell-major)
__global__ void init_state_kernel(const float* __restrict__ code,
                                  _Float16* __restrict__ h0,
                                  float* __restrict__ c0) {
  const int idx  = blockIdx.x * 256 + threadIdx.x;  // 4*BH total
  const int cell = idx >> 18;
  const int rem  = idx & (BH - 1);
  const int b    = rem >> 8;
  const int j    = rem & 255;
  const size_t base = (size_t)b * 2048 + (size_t)cell * 256 + j;
  h0[idx] = (_Float16)code[base];
  c0[idx] = code[base + 1024];
}

__global__ void zero_h_kernel(_Float16* __restrict__ z) {
  const int idx = blockIdx.x * 256 + threadIdx.x;  // BH total
  z[idx] = (_Float16)0.f;
}

// ---------------------------------------------------------------------------
// Host launcher
// ---------------------------------------------------------------------------
extern "C" void kernel_launch(void* const* d_in, const int* in_sizes, int n_in,
                              void* d_out, int out_size, void* d_ws, size_t ws_size,
                              hipStream_t stream) {
  (void)in_sizes; (void)n_in; (void)out_size; (void)ws_size;

  const float* code = (const float*)d_in[0];
  // d_in[1] ("x") is unused by the reference computation.
  const float* Wih0 = (const float*)d_in[2];
  const float* Whh0 = (const float*)d_in[3];
  const float* bih0 = (const float*)d_in[4];
  const float* bhh0 = (const float*)d_in[5];
  const float* Wih1 = (const float*)d_in[6];
  const float* Whh1 = (const float*)d_in[7];
  const float* bih1 = (const float*)d_in[8];
  const float* bhh1 = (const float*)d_in[9];

  // Workspace layout (~14.5 MB)
  char* ws = (char*)d_ws;
  _Float16* Wc0  = (_Float16*)ws;                       // 2*1024*768 f16
  _Float16* Wc1  = Wc0 + 2 * 1024 * KTOT;               // + 3,145,728 B
  float*    bia0 = (float*)(ws + 6291456);              // 2*1024 f32
  float*    bia1 = bia0 + 2048;
  _Float16* hA   = (_Float16*)(ws + 6291456 + 16384);   // 4*BH f16
  _Float16* hB   = hA + 4 * BH;                         // 4*BH f16
  float*    cst  = (float*)(ws + 6291456 + 16384 + (size_t)2 * 4 * BH * 2);
  _Float16* zros = (_Float16*)((char*)cst + (size_t)4 * BH * 4);

  prep_w_kernel<<<6144, 256, 0, stream>>>(Wih0, Whh0, Wc0);
  prep_w_kernel<<<6144, 256, 0, stream>>>(Wih1, Whh1, Wc1);
  prep_b_kernel<<<8, 256, 0, stream>>>(bih0, bhh0, bia0);
  prep_b_kernel<<<8, 256, 0, stream>>>(bih1, bhh1, bia1);
  init_state_kernel<<<4096, 256, 0, stream>>>(code, hA, cst);
  zero_h_kernel<<<1024, 256, 0, stream>>>(zros);

  float* out = (float*)d_out;
  _Float16* P = hA;  // state entering the step
  _Float16* Q = hB;  // state produced by the step

  const dim3 grid(16, 8, 2), blk(256);
  for (int t = 0; t < NSTEP; ++t) {
    const _Float16* x0 = t ? (P + (size_t)2 * BH) : zros;  // prev y lo = h[2]
    const _Float16* x1 = t ? (P + (size_t)3 * BH) : zros;  // prev y hi = h[3]
    // layer 0 (cells 0,1): reads x0|x1|P[d], writes Q[0..1], c[0..1]
    lstm_layer_kernel<<<grid, blk, 0, stream>>>(
        x0, x1, P, Q, cst, Wc0, bia0, nullptr);
    // layer 1 (cells 2,3): reads Q[0]|Q[1]|P[2+d], writes Q[2..3], c[2..3], y
    lstm_layer_kernel<<<grid, blk, 0, stream>>>(
        Q, Q + BH, P + (size_t)2 * BH, Q + (size_t)2 * BH,
        cst + (size_t)2 * BH, Wc1, bia1, out + (size_t)t * NF_);
    _Float16* tmp = P; P = Q; Q = tmp;
  }
}
